// ConvGraphNet_7756710936771
// MI455X (gfx1250) — matrix operand order repaired and checked by reading, not compiled, not run on hardware
//
#include <hip/hip_runtime.h>
#include <hip/hip_bf16.h>

// ---------- types ----------
typedef __attribute__((ext_vector_type(16))) __bf16         v16bf;
typedef __attribute__((ext_vector_type(8)))  float          v8f;
typedef __attribute__((ext_vector_type(16))) unsigned short v16u;
typedef __attribute__((ext_vector_type(8)))  unsigned short v8u;

// ---------- bf16 helpers (round-to-nearest-even) ----------
__device__ __forceinline__ unsigned short f2bf(float f) {
    unsigned int u = __float_as_uint(f);
    u += 0x7FFFu + ((u >> 16) & 1u);
    return (unsigned short)(u >> 16);
}
__device__ __forceinline__ float bf2f(unsigned short h) {
    return __uint_as_float(((unsigned int)h) << 16);
}

// ---------- degree / norm pre-pass ----------
__global__ void k_init_deg(float* deg, int N) {
    int i = blockIdx.x * blockDim.x + threadIdx.x;
    if (i < N) deg[i] = 1.0f;   // self-loop contributes 1
}
__global__ void k_deg_accum(const int* __restrict__ dst, float* deg, int E) {
    int e = blockIdx.x * blockDim.x + threadIdx.x;
    if (e < E) atomicAdd(&deg[dst[e]], 1.0f);
}
__global__ void k_dinv(float* deg, int N) {
    int i = blockIdx.x * blockDim.x + threadIdx.x;
    if (i < N) deg[i] = rsqrtf(deg[i]);   // deg >= 1 always
}
__global__ void k_norm(const int* __restrict__ src, const int* __restrict__ dst,
                       const float* __restrict__ dinv, float* nrm, int E) {
    int e = blockIdx.x * blockDim.x + threadIdx.x;
    if (e < E) nrm[e] = dinv[src[e]] * dinv[dst[e]];
}

// ---------- activation f32 -> bf16 ----------
__global__ void k_x2bf(const float* __restrict__ x, unsigned short* __restrict__ xb, int n) {
    int i = blockIdx.x * blockDim.x + threadIdx.x;
    if (i < n) xb[i] = f2bf(x[i]);
}

// ---------- pack W into WMMA B-fragment order ----------
// Tile (32K x 16N). Per tile: 32 lanes x 16 bf16, lane = ((k%32)/8)%2*16 + n%16,
// element e = k%8 + (k%32)/16*8  -> every wave operand load is a contiguous 32B per lane.
__global__ void k_packW(const float* __restrict__ W, unsigned short* __restrict__ Wpk,
                        int F_in, int F_out) {
    int i = blockIdx.x * blockDim.x + threadIdx.x;
    if (i >= F_in * F_out) return;
    int k = i / F_out, n = i - k * F_out;
    int kt = k >> 5, kl = k & 31;
    int khalf = (kl >> 3) & 1, e = (kl & 7) + ((kl >> 4) << 3);
    int lane = (khalf << 4) + (n & 15);
    int nt = n >> 4;
    size_t tile = (size_t)kt * (size_t)(F_out >> 4) + (size_t)nt;
    Wpk[(tile * 32 + lane) * 16 + e] = f2bf(W[i]);
}

// ---------- WMMA GEMM: H[N,F_out] = Xbf[N,F_in] @ Wpk ----------
// One wave computes a 16x64 output strip: A fragment loaded once, 4 WMMAs.
__global__ void k_gemm(const unsigned short* __restrict__ Xbf,
                       const unsigned short* __restrict__ Wpk,
                       float* __restrict__ H,
                       int N, int F_in, int F_out) {
    const int lane = threadIdx.x & 31;
    const int wave = (blockIdx.x * blockDim.x + threadIdx.x) >> 5;
    const int colGroups = F_out >> 6;          // 64-col groups
    const int rowTiles  = N >> 4;
    const int rowTile = wave / colGroups;
    const int cg      = wave - rowTile * colGroups;
    if (rowTile >= rowTiles) return;

    // A layout (16-bit 16x32): lane<16 -> row=lane, K in {0..7,16..23};
    // lane>=16 -> row=lane-16, K in {8..15,24..31}.
    const int row = (rowTile << 4) + (lane & 15);
    const int kh  = lane >> 4;
    const unsigned short* arow = Xbf + (size_t)row * (size_t)F_in;

    v8f acc0 = {}; v8f acc1 = {}; v8f acc2 = {}; v8f acc3 = {};
    const int tilesPerK = F_out >> 4;
    const int tbase = cg << 2;

    for (int k0 = 0; k0 < F_in; k0 += 32) {
        v8u alo = *(const v8u*)(arow + k0 + kh * 8);
        v8u ahi = *(const v8u*)(arow + k0 + 16 + kh * 8);
        v16u au = __builtin_shufflevector(alo, ahi,
                    0,1,2,3,4,5,6,7,8,9,10,11,12,13,14,15);
        v16bf a = __builtin_bit_cast(v16bf, au);

        const unsigned short* wk =
            Wpk + (size_t)(k0 >> 5) * (size_t)tilesPerK * 512 + (size_t)lane * 16;
        v16bf b0 = __builtin_bit_cast(v16bf, *(const v16u*)(wk + (size_t)(tbase + 0) * 512));
        v16bf b1 = __builtin_bit_cast(v16bf, *(const v16u*)(wk + (size_t)(tbase + 1) * 512));
        v16bf b2 = __builtin_bit_cast(v16bf, *(const v16u*)(wk + (size_t)(tbase + 2) * 512));
        v16bf b3 = __builtin_bit_cast(v16bf, *(const v16u*)(wk + (size_t)(tbase + 3) * 512));

        acc0 = __builtin_amdgcn_wmma_f32_16x16x32_bf16(false, a, false, b0, (short)0, acc0, false, false);
        acc1 = __builtin_amdgcn_wmma_f32_16x16x32_bf16(false, a, false, b1, (short)0, acc1, false, false);
        acc2 = __builtin_amdgcn_wmma_f32_16x16x32_bf16(false, a, false, b2, (short)0, acc2, false, false);
        acc3 = __builtin_amdgcn_wmma_f32_16x16x32_bf16(false, a, false, b3, (short)0, acc3, false, false);
    }

    // C/D layout: lanes 0-15 N=lane, M=r; lanes 16-31 N=lane-16, M=8+r.
    const int outRow  = (rowTile << 4) + (lane >> 4) * 8;
    const int colBase = (cg << 6) + (lane & 15);
    float* hp = H + (size_t)outRow * (size_t)F_out + colBase;
#pragma unroll
    for (int r = 0; r < 8; ++r) {
        hp[(size_t)r * F_out +  0] = acc0[r];
        hp[(size_t)r * F_out + 16] = acc1[r];
        hp[(size_t)r * F_out + 32] = acc2[r];
        hp[(size_t)r * F_out + 48] = acc3[r];
    }
}

// ---------- edge scatter: AGG[dst] += norm * H[src] ----------
__global__ void k_scatter(const int* __restrict__ src, const int* __restrict__ dst,
                          const float* __restrict__ nrm, const float* __restrict__ H,
                          float* __restrict__ AGG, int E, int F) {
    int gid = blockIdx.x * blockDim.x + threadIdx.x;
    int chunks = F >> 2;
    int e = gid / chunks;
    if (e >= E) return;
    int c = (gid - e * chunks) << 2;
    int s = src[e], d = dst[e];
    float w = nrm[e];
    const float* hs = H + (size_t)s * F + c;
    float* ad = AGG + (size_t)d * F + c;
    atomicAdd(ad + 0, w * hs[0]);
    atomicAdd(ad + 1, w * hs[1]);
    atomicAdd(ad + 2, w * hs[2]);
    atomicAdd(ad + 3, w * hs[3]);
}

// ---------- epilogue: + self-loop + bias, ReLU, -> bf16 next-layer input ----------
__global__ void k_epilogue_bf(const float* __restrict__ AGG, const float* __restrict__ H,
                              const float* __restrict__ dinv, const float* __restrict__ bias,
                              unsigned short* __restrict__ Xbf, int N, int F) {
    int i = blockIdx.x * blockDim.x + threadIdx.x;
    if (i >= N * F) return;
    int n = i / F, f = i - n * F;
    float dn = dinv[n];
    float v = AGG[i] + dn * dn * H[i] + bias[f];
    Xbf[i] = f2bf(v > 0.0f ? v : 0.0f);
}

// ---------- layer 5 (64 -> 1) ----------
__global__ void k_gemv64(const unsigned short* __restrict__ Xbf,
                         const float* __restrict__ W5, float* __restrict__ h, int N) {
    int n = blockIdx.x * blockDim.x + threadIdx.x;
    if (n >= N) return;
    const unsigned short* xr = Xbf + (size_t)n * 64;
    float acc = 0.0f;
#pragma unroll
    for (int k = 0; k < 64; ++k) acc += bf2f(xr[k]) * W5[k];
    h[n] = acc;
}
__global__ void k_scatter1(const int* __restrict__ src, const int* __restrict__ dst,
                           const float* __restrict__ nrm, const float* __restrict__ h,
                           float* __restrict__ agg, int E) {
    int e = blockIdx.x * blockDim.x + threadIdx.x;
    if (e < E) atomicAdd(&agg[dst[e]], nrm[e] * h[src[e]]);
}
__global__ void k_final(const float* __restrict__ agg, const float* __restrict__ h,
                        const float* __restrict__ dinv, const float* __restrict__ b5,
                        float* __restrict__ out, int N) {
    int n = blockIdx.x * blockDim.x + threadIdx.x;
    if (n >= N) return;
    float dn = dinv[n];
    float v = agg[n] + dn * dn * h[n] + b5[0];
    out[n] = v > 0.0f ? v : 0.0f;
}

// ---------- launcher ----------
extern "C" void kernel_launch(void* const* d_in, const int* in_sizes, int n_in,
                              void* d_out, int out_size, void* d_ws, size_t ws_size,
                              hipStream_t stream) {
    const float* x  = (const float*)d_in[0];
    const int*   ei = (const int*)d_in[1];
    const float* W[5] = {(const float*)d_in[2], (const float*)d_in[4], (const float*)d_in[6],
                         (const float*)d_in[8], (const float*)d_in[10]};
    const float* B[5] = {(const float*)d_in[3], (const float*)d_in[5], (const float*)d_in[7],
                         (const float*)d_in[9], (const float*)d_in[11]};
    const int DIMS[6] = {128, 1024, 512, 256, 64, 1};
    const int N = in_sizes[0] / DIMS[0];
    const int E = in_sizes[1] / 2;
    const int* srcI = ei;
    const int* dstI = ei + E;

    // workspace carve (256B aligned)
    char* p = (char*)d_ws;
    auto carve = [&](size_t bytes) -> char* {
        char* r = p;
        p += (bytes + 255) & ~(size_t)255;
        return r;
    };
    float*          dinv = (float*)carve((size_t)N * 4);
    float*          nrm  = (float*)carve((size_t)E * 4);
    unsigned short* Xbf  = (unsigned short*)carve((size_t)N * 1024 * 2);
    float*          H    = (float*)carve((size_t)N * 1024 * 4);
    float*          AGG  = (float*)carve((size_t)N * 1024 * 4);
    unsigned short* Wpk  = (unsigned short*)carve((size_t)1024 * 1024 * 2);

    const int T = 256;
    auto blk = [&](long long n) { return (int)((n + T - 1) / T); };

    // degree / norm pre-pass (reused by every layer)
    k_init_deg<<<blk(N), T, 0, stream>>>(dinv, N);
    k_deg_accum<<<blk(E), T, 0, stream>>>(dstI, dinv, E);
    k_dinv<<<blk(N), T, 0, stream>>>(dinv, N);
    k_norm<<<blk(E), T, 0, stream>>>(srcI, dstI, dinv, nrm, E);
    k_x2bf<<<blk((long long)N * 128), T, 0, stream>>>(x, Xbf, N * 128);

    // layers 1..4: WMMA GEMM + scatter + fused epilogue
    for (int L = 0; L < 4; ++L) {
        int Fi = DIMS[L], Fo = DIMS[L + 1];
        k_packW<<<blk((long long)Fi * Fo), T, 0, stream>>>(W[L], Wpk, Fi, Fo);
        long long waves = (long long)(N / 16) * (Fo / 64);
        k_gemm<<<blk(waves * 32), T, 0, stream>>>(Xbf, Wpk, H, N, Fi, Fo);
        hipMemsetAsync(AGG, 0, (size_t)N * Fo * 4, stream);
        k_scatter<<<blk((long long)E * (Fo / 4)), T, 0, stream>>>(srcI, dstI, nrm, H, AGG, E, Fo);
        k_epilogue_bf<<<blk((long long)N * Fo), T, 0, stream>>>(AGG, H, dinv, B[L], Xbf, N, Fo);
    }

    // layer 5: 64 -> 1
    k_gemv64<<<blk(N), T, 0, stream>>>(Xbf, W[4], H, N);
    hipMemsetAsync(AGG, 0, (size_t)N * 4, stream);
    k_scatter1<<<blk(E), T, 0, stream>>>(srcI, dstI, nrm, H, AGG, E);
    k_final<<<blk(N), T, 0, stream>>>(AGG, H, dinv, B[4], (float*)d_out, N);
}
